// MultiHeadAttention_77137612636349
// MI455X (gfx1250) — matrix operand order repaired and checked by reading, not compiled, not run on hardware
//
#include <hip/hip_runtime.h>

#define B_  4
#define S_  2048
#define D_  1024
#define H_  16
#define DH_ 64
#define BS_ (B_ * S_)
#define EPS_ 1e-5f

typedef __attribute__((ext_vector_type(16))) __bf16 v16bf;
typedef __attribute__((ext_vector_type(8)))  float  v8f;

union AB16 { v16bf v; unsigned u[8]; };

__device__ __forceinline__ unsigned short f2bf(float f) {
  union { float f; unsigned u; } c; c.f = f;
  unsigned r = c.u + 0x7FFFu + ((c.u >> 16) & 1u);   // round-to-nearest-even
  return (unsigned short)(r >> 16);
}

__device__ __forceinline__ v8f wmma_bf16(v16bf a, v16bf b, v8f c) {
  return __builtin_amdgcn_wmma_f32_16x16x32_bf16(false, a, false, b, (short)0, c, false, false);
}

__device__ __forceinline__ v8f vzero8() { v8f z = {0,0,0,0,0,0,0,0}; return z; }

// A-matrix 16x32 bf16 fragment from row-major src (rowStride elements).
// Lane L: M = L%16, g = L/16. VGPR v holds K pair: v<4 -> 2v+8g, else 16+2(v-4)+8g.
__device__ __forceinline__ v16bf load_a_frag(const unsigned short* __restrict__ src,
                                             int rowStride, int m0, int k0) {
  int lane = threadIdx.x & 31;
  int M = lane & 15, g = lane >> 4;
  const unsigned short* p = src + (size_t)(m0 + M) * rowStride + k0;
  AB16 f;
#pragma unroll
  for (int v = 0; v < 8; ++v) {
    int kb = (v < 4) ? (2 * v + 8 * g) : (16 + 2 * (v - 4) + 8 * g);
    f.u[v] = *(const unsigned*)(p + kb);
  }
  return f.v;
}

// B-matrix 32x16 bf16 fragment. src is "N-major": element (K,N) at src[n*kStride + k].
// Lane L: N = L%16, g = L/16. VGPR v holds K pair: v<4 -> 2v+8g, else 16+2(v-4)+8g.
__device__ __forceinline__ v16bf load_b_frag(const unsigned short* __restrict__ src,
                                             int kStride, int n0, int k0) {
  int lane = threadIdx.x & 31;
  int N = lane & 15, g = lane >> 4;
  const unsigned short* p = src + (size_t)(n0 + N) * kStride + k0 + 8 * g;
  AB16 f;
#pragma unroll
  for (int v = 0; v < 8; ++v) f.u[v] = *(const unsigned*)(p + ((v < 4) ? 2 * v : 16 + 2 * (v - 4)));
  return f.v;
}

__device__ __forceinline__ float half_max(float x) {
  x = fmaxf(x, __shfl_xor(x, 1, 32));
  x = fmaxf(x, __shfl_xor(x, 2, 32));
  x = fmaxf(x, __shfl_xor(x, 4, 32));
  x = fmaxf(x, __shfl_xor(x, 8, 32));
  return x;
}
__device__ __forceinline__ float half_sum(float x) {
  x += __shfl_xor(x, 1, 32);
  x += __shfl_xor(x, 2, 32);
  x += __shfl_xor(x, 4, 32);
  x += __shfl_xor(x, 8, 32);
  return x;
}

// ---------------- conversion kernels ----------------
__global__ void cvt_f32_bf16(const float* __restrict__ src, unsigned short* __restrict__ dst, int n) {
  for (int i = blockIdx.x * blockDim.x + threadIdx.x; i < n; i += gridDim.x * blockDim.x)
    dst[i] = f2bf(src[i]);
}

// W [H][D][DH] f32 -> Wt [H][DH][D] bf16 (K contiguous for B fragments)
__global__ void cvt_w_hdk(const float* __restrict__ w, unsigned short* __restrict__ wt, int n) {
  for (int i = blockIdx.x * blockDim.x + threadIdx.x; i < n; i += gridDim.x * blockDim.x) {
    int h = i / (DH_ * D_);
    int r = i - h * (DH_ * D_);
    int nn = r / D_;
    int kk = r - nn * D_;
    wt[i] = f2bf(w[((size_t)h * D_ + kk) * DH_ + nn]);
  }
}

// Wo [in][out] f32 -> Wot [out][in] bf16
__global__ void cvt_wo(const float* __restrict__ w, unsigned short* __restrict__ wt, int n) {
  for (int i = blockIdx.x * blockDim.x + threadIdx.x; i < n; i += gridDim.x * blockDim.x) {
    int o = i / D_;
    int ii = i - o * D_;
    wt[i] = f2bf(w[(size_t)ii * D_ + o]);
  }
}

// ---------------- QKV projection ----------------
// grid: (BS/32, H, 3). One wave computes a 32 x 64 tile of q/k/v for one head,
// reusing each weight B-fragment across two WMMAs.
__global__ __launch_bounds__(32) void qkv_gemm(const unsigned short* __restrict__ xb,
                                               const unsigned short* __restrict__ wqt,
                                               const unsigned short* __restrict__ wkt,
                                               const unsigned short* __restrict__ wvt,
                                               unsigned short* __restrict__ q,
                                               unsigned short* __restrict__ k,
                                               unsigned short* __restrict__ vt) {
  const int mt = blockIdx.x, h = blockIdx.y, which = blockIdx.z;
  const unsigned short* wt = (which == 0) ? wqt : (which == 1) ? wkt : wvt;
  wt += (size_t)h * DH_ * D_;  // [DH][D]

  v8f acc[2][4];
#pragma unroll
  for (int u = 0; u < 2; ++u)
#pragma unroll
    for (int t = 0; t < 4; ++t) acc[u][t] = vzero8();

  for (int k0 = 0; k0 < D_; k0 += 32) {
    v16bf a0 = load_a_frag(xb, D_, mt * 32,      k0);
    v16bf a1 = load_a_frag(xb, D_, mt * 32 + 16, k0);
#if __has_builtin(__builtin_prefetch)
    {
      int lane = threadIdx.x & 31;
      __builtin_prefetch((const void*)(xb + (size_t)(mt * 32 + (lane & 15)) * D_ + k0 + 64), 0, 1);
      __builtin_prefetch((const void*)(wt + (size_t)(lane & 15) * D_ + k0 + 64), 0, 1);
    }
#endif
#pragma unroll
    for (int t = 0; t < 4; ++t) {
      v16bf bw = load_b_frag(wt, D_, t * 16, k0);
      acc[0][t] = wmma_bf16(a0, bw, acc[0][t]);
      acc[1][t] = wmma_bf16(a1, bw, acc[1][t]);
    }
  }
  const int lane = threadIdx.x & 31, N = lane & 15, g = lane >> 4;
#pragma unroll
  for (int u = 0; u < 2; ++u) {
#pragma unroll
    for (int r = 0; r < 8; ++r) {
      int row = mt * 32 + u * 16 + r + 8 * g;  // global token index (tiles never cross batch)
      int b = row / S_, s = row - b * S_;
      size_t bh = (size_t)b * H_ + h;
#pragma unroll
      for (int t = 0; t < 4; ++t) {
        unsigned short val = f2bf(acc[u][t][r]);
        if (which == 2)       vt[(bh * DH_ + t * 16 + N) * S_ + s] = val;   // [b][h][dh][s]
        else if (which == 1)  k [(bh * S_ + s) * DH_ + t * 16 + N] = val;   // [b][h][s][dh]
        else                  q [(bh * S_ + s) * DH_ + t * 16 + N] = val;
      }
    }
  }
}

// ---------------- flash attention ----------------
// grid: (S/32, B*H). One wave owns 32 query rows (two 16-row C tiles), streams KV
// in steps of 32; K/V B-fragments are reused across both query tiles.
__global__ __launch_bounds__(32) void flash_attn(const unsigned short* __restrict__ q,
                                                 const unsigned short* __restrict__ k,
                                                 const unsigned short* __restrict__ vt,
                                                 unsigned short* __restrict__ ctx) {
  const int qt = blockIdx.x, bh = blockIdx.y;
  const int b = bh / H_, h = bh - b * H_;
  const unsigned short* qh = q  + (size_t)bh * S_ * DH_;
  const unsigned short* kh = k  + (size_t)bh * S_ * DH_;
  const unsigned short* vh = vt + (size_t)bh * DH_ * S_;   // [DH][S]
  const float inv_scale = 0.125f;                          // 1/sqrt(64)

  const int lane = threadIdx.x & 31, N = lane & 15, g = lane >> 4;

  v16bf aQ[2][2];
#pragma unroll
  for (int u = 0; u < 2; ++u) {
    aQ[u][0] = load_a_frag(qh, DH_, qt * 32 + u * 16, 0);
    aQ[u][1] = load_a_frag(qh, DH_, qt * 32 + u * 16, 32);
  }

  v8f acc[2][4];
  float m_[2][8], l_[2][8];
#pragma unroll
  for (int u = 0; u < 2; ++u) {
#pragma unroll
    for (int t = 0; t < 4; ++t) acc[u][t] = vzero8();
#pragma unroll
    for (int r = 0; r < 8; ++r) { m_[u][r] = -1e30f; l_[u][r] = 0.0f; }
  }

  __shared__ unsigned short pL[2][16 * 32];  // staged probabilities per q tile

  for (int kv = 0; kv < S_; kv += 32) {
    // ---- K fragments loaded once, shared by both q tiles ----
    v16bf bK00 = load_b_frag(kh, DH_, kv,       0);
    v16bf bK01 = load_b_frag(kh, DH_, kv,      32);
    v16bf bK10 = load_b_frag(kh, DH_, kv + 16,  0);
    v16bf bK11 = load_b_frag(kh, DH_, kv + 16, 32);

    __syncthreads();
#pragma unroll
    for (int u = 0; u < 2; ++u) {
      v8f s0 = vzero8(), s1 = vzero8();
      s0 = wmma_bf16(aQ[u][0], bK00, s0);
      s0 = wmma_bf16(aQ[u][1], bK01, s0);
      s1 = wmma_bf16(aQ[u][0], bK10, s1);
      s1 = wmma_bf16(aQ[u][1], bK11, s1);

      // ---- online softmax (per C-tile row r; a row lives in one 16-lane half) ----
      float p0[8], p1[8];
#pragma unroll
      for (int r = 0; r < 8; ++r) {
        float sa = s0[r] * inv_scale, sb = s1[r] * inv_scale;
        float mx = half_max(fmaxf(sa, sb));
        float mn = fmaxf(m_[u][r], mx);
        float al = __expf(m_[u][r] - mn);
        p0[r] = __expf(sa - mn);
        p1[r] = __expf(sb - mn);
        l_[u][r] = l_[u][r] * al + half_sum(p0[r] + p1[r]);
        m_[u][r] = mn;
#pragma unroll
        for (int t = 0; t < 4; ++t) acc[u][t][r] *= al;
      }

      // ---- stage P (16x32 bf16) through LDS for A-layout re-read ----
#pragma unroll
      for (int r = 0; r < 8; ++r) {
        int row = r + 8 * g;
        pL[u][row * 32 + N]      = f2bf(p0[r]);
        pL[u][row * 32 + 16 + N] = f2bf(p1[r]);
      }
    }
    __syncthreads();

    AB16 pf[2];
#pragma unroll
    for (int u = 0; u < 2; ++u) {
      const unsigned short* pp = &pL[u][(lane & 15) * 32];
#pragma unroll
      for (int v = 0; v < 8; ++v) {
        int kb = (v < 4) ? (2 * v + 8 * g) : (16 + 2 * (v - 4) + 8 * g);
        pf[u].u[v] = *(const unsigned*)(pp + kb);
      }
    }

    // ---- ctx += P @ V; each V fragment reused by both q tiles ----
#pragma unroll
    for (int t = 0; t < 4; ++t) {
      v16bf bV = load_b_frag(vh, S_, t * 16, kv);
      acc[0][t] = wmma_bf16(pf[0].v, bV, acc[0][t]);
      acc[1][t] = wmma_bf16(pf[1].v, bV, acc[1][t]);
    }
  }

  // ---- normalize and store concat-head layout [b][s][h*DH+dh] ----
#pragma unroll
  for (int u = 0; u < 2; ++u) {
#pragma unroll
    for (int r = 0; r < 8; ++r) {
      float inv_l = 1.0f / l_[u][r];
      int row = qt * 32 + u * 16 + r + 8 * g;
      size_t base = ((size_t)b * S_ + row) * D_ + h * DH_;
#pragma unroll
      for (int t = 0; t < 4; ++t)
        ctx[base + t * 16 + N] = f2bf(acc[u][t][r] * inv_l);
    }
  }
}

// ---------------- output projection + bias + residual ----------------
// grid: (BS/32, D/64). One wave: 32 x 64 output tile, K = D; B-fragments reused 2x.
__global__ __launch_bounds__(32) void proj_residual(const unsigned short* __restrict__ ctx,
                                                    const unsigned short* __restrict__ wot,
                                                    const float* __restrict__ x,
                                                    const float* __restrict__ bo,
                                                    float* __restrict__ out) {
  const int mt = blockIdx.x, nt = blockIdx.y;
  v8f acc[2][4];
#pragma unroll
  for (int u = 0; u < 2; ++u)
#pragma unroll
    for (int t = 0; t < 4; ++t) acc[u][t] = vzero8();

  for (int k0 = 0; k0 < D_; k0 += 32) {
    v16bf a0 = load_a_frag(ctx, D_, mt * 32,      k0);
    v16bf a1 = load_a_frag(ctx, D_, mt * 32 + 16, k0);
#if __has_builtin(__builtin_prefetch)
    {
      int lane = threadIdx.x & 31;
      __builtin_prefetch((const void*)(ctx + (size_t)(mt * 32 + (lane & 15)) * D_ + k0 + 64), 0, 1);
      __builtin_prefetch((const void*)(wot + (size_t)(nt * 64 + (lane & 15)) * D_ + k0 + 64), 0, 1);
    }
#endif
#pragma unroll
    for (int t = 0; t < 4; ++t) {
      v16bf bw = load_b_frag(wot, D_, nt * 64 + t * 16, k0);
      acc[0][t] = wmma_bf16(a0, bw, acc[0][t]);
      acc[1][t] = wmma_bf16(a1, bw, acc[1][t]);
    }
  }
  const int lane = threadIdx.x & 31, N = lane & 15, g = lane >> 4;
#pragma unroll
  for (int u = 0; u < 2; ++u) {
#pragma unroll
    for (int r = 0; r < 8; ++r) {
      int row = mt * 32 + u * 16 + r + 8 * g;
#pragma unroll
      for (int t = 0; t < 4; ++t) {
        int col = nt * 64 + t * 16 + N;
        size_t idx = (size_t)row * D_ + col;
        out[idx] = acc[u][t][r] + bo[col] + x[idx];
      }
    }
  }
}

// ---------------- layernorm (in-place on d_out) ----------------
__global__ __launch_bounds__(256) void layernorm_inplace(float* __restrict__ out,
                                                         const float* __restrict__ gamma,
                                                         const float* __restrict__ beta) {
  __shared__ float redS[8], redQ[8];
  size_t base = (size_t)blockIdx.x * D_;
  float v[4], s = 0.0f, ss = 0.0f;
#pragma unroll
  for (int i = 0; i < 4; ++i) {
    v[i] = out[base + threadIdx.x + i * 256];
    s += v[i]; ss += v[i] * v[i];
  }
#pragma unroll
  for (int m = 1; m < 32; m <<= 1) { s += __shfl_xor(s, m, 32); ss += __shfl_xor(ss, m, 32); }
  int w = threadIdx.x >> 5;
  if ((threadIdx.x & 31) == 0) { redS[w] = s; redQ[w] = ss; }
  __syncthreads();
  float ts = 0.0f, tq = 0.0f;
#pragma unroll
  for (int j = 0; j < 8; ++j) { ts += redS[j]; tq += redQ[j]; }
  float mu = ts * (1.0f / D_);
  float var = tq * (1.0f / D_) - mu * mu;
  float inv = rsqrtf(var + EPS_);
#pragma unroll
  for (int i = 0; i < 4; ++i) {
    int col = threadIdx.x + i * 256;
    out[base + col] = (v[i] - mu) * inv * gamma[col] + beta[col];
  }
}

// ---------------- launcher ----------------
extern "C" void kernel_launch(void* const* d_in, const int* in_sizes, int n_in,
                              void* d_out, int out_size, void* d_ws, size_t ws_size,
                              hipStream_t stream) {
  (void)in_sizes; (void)n_in; (void)out_size; (void)ws_size;
  const float* x     = (const float*)d_in[0];
  const float* Wq    = (const float*)d_in[1];
  const float* Wk    = (const float*)d_in[2];
  const float* Wv    = (const float*)d_in[3];
  const float* Wo    = (const float*)d_in[4];
  const float* bo    = (const float*)d_in[5];
  const float* gamma = (const float*)d_in[6];
  const float* beta  = (const float*)d_in[7];
  float* out = (float*)d_out;

  char* ws = (char*)d_ws;
  size_t off = 0;
  auto take = [&](size_t bytes) -> void* {
    void* p = ws + off;
    off = (off + bytes + 255) & ~(size_t)255;
    return p;
  };
  unsigned short* xb  = (unsigned short*)take((size_t)BS_ * D_ * 2);
  unsigned short* wqt = (unsigned short*)take((size_t)H_ * DH_ * D_ * 2);
  unsigned short* wkt = (unsigned short*)take((size_t)H_ * DH_ * D_ * 2);
  unsigned short* wvt = (unsigned short*)take((size_t)H_ * DH_ * D_ * 2);
  unsigned short* wot = (unsigned short*)take((size_t)D_ * D_ * 2);
  unsigned short* qB  = (unsigned short*)take((size_t)B_ * H_ * S_ * DH_ * 2);
  unsigned short* kB  = (unsigned short*)take((size_t)B_ * H_ * S_ * DH_ * 2);
  unsigned short* vtB = (unsigned short*)take((size_t)B_ * H_ * DH_ * S_ * 2);
  unsigned short* ctx = (unsigned short*)take((size_t)BS_ * D_ * 2);

  // 1) precision conversion + weight re-layout
  cvt_f32_bf16<<<2048, 256, 0, stream>>>(x, xb, BS_ * D_);
  cvt_w_hdk  <<<1024, 256, 0, stream>>>(Wq, wqt, H_ * DH_ * D_);
  cvt_w_hdk  <<<1024, 256, 0, stream>>>(Wk, wkt, H_ * DH_ * D_);
  cvt_w_hdk  <<<1024, 256, 0, stream>>>(Wv, wvt, H_ * DH_ * D_);
  cvt_wo     <<<1024, 256, 0, stream>>>(Wo, wot, D_ * D_);

  // 2) QKV projections (WMMA, 32-row tiles)
  qkv_gemm<<<dim3(BS_ / 32, H_, 3), 32, 0, stream>>>(xb, wqt, wkt, wvt, qB, kB, vtB);

  // 3) flash attention (WMMA + LDS-staged softmax, 32-row q tiles)
  flash_attn<<<dim3(S_ / 32, B_ * H_), 32, 0, stream>>>(qB, kB, vtB, ctx);

  // 4) output projection + bias + residual (WMMA, 32-row tiles)
  proj_residual<<<dim3(BS_ / 32, D_ / 64), 32, 0, stream>>>(ctx, wot, x, bo, out);

  // 5) layernorm in place
  layernorm_inplace<<<BS_, 256, 0, stream>>>(out, gamma, beta);
}
